// Net_9964324127007
// MI455X (gfx1250) — matrix operand order repaired and checked by reading, compiled, not run on hardware
//
#include <hip/hip_runtime.h>
#include <hip/hip_bf16.h>
#include <math.h>

typedef __attribute__((ext_vector_type(16))) _Float16 v16h;
typedef __attribute__((ext_vector_type(8)))  float    v8f;

#define BGR   256        // num graphs
#define NPG0  128        // nodes per graph (layer 1 input)
#define EPG   1024
#define NEDGE (BGR*EPG)  // 262144
#define DFEAT 128
#define N0    (BGR*NPG0) // 32768

// ---------------- WMMA GEMM: C[M,N] = A[M,K] @ B[K,N] (+bias, relu) ---------
// fp32 in/out, f16 multiply, f32 accumulate. One wave computes one 16x16 tile.
// K compile-time (multiple of 32) so the K-loop fully unrolls into a chain of
// v_wmma_f32_16x16x32_f16. Requires gridDim = (M/16, N/16).
template <int K, bool BIAS, bool RELU>
__global__ __launch_bounds__(32) void k_wmma_gemm(
    const float* __restrict__ A, const float* __restrict__ Bm,
    float* __restrict__ C, int N, const float* __restrict__ bias)
{
  const int tile_m = blockIdx.x;
  const int tile_n = blockIdx.y;
  const int lane   = threadIdx.x;        // 0..31
  const int hlf    = lane >> 4;          // 0/1
  const int l16    = lane & 15;
  const int rowA   = tile_m * 16 + l16;  // A row for this lane
  const int colB   = tile_n * 16 + l16;  // B column / C column

  v8f acc = {};
#pragma unroll
  for (int k0 = 0; k0 < K; k0 += 32) {
    // ---- A fragment (16x32 f16): lane holds M=l16.
    //      elems 0..7  -> K = k0 + 8*hlf + e          (contiguous, 32B aligned)
    //      elems 8..15 -> K = k0 + 16 + 8*hlf + (e-8) (contiguous, 32B aligned)
    const float4* ap = reinterpret_cast<const float4*>(
        A + (size_t)rowA * K + k0 + hlf * 8);
    float4 a0 = ap[0], a1 = ap[1];   // K run 1 (8 floats)
    float4 a2 = ap[4], a3 = ap[5];   // K run 2 (+16 floats)
    v16h a;
    a[0]  = (_Float16)a0.x; a[1]  = (_Float16)a0.y;
    a[2]  = (_Float16)a0.z; a[3]  = (_Float16)a0.w;
    a[4]  = (_Float16)a1.x; a[5]  = (_Float16)a1.y;
    a[6]  = (_Float16)a1.z; a[7]  = (_Float16)a1.w;
    a[8]  = (_Float16)a2.x; a[9]  = (_Float16)a2.y;
    a[10] = (_Float16)a2.z; a[11] = (_Float16)a2.w;
    a[12] = (_Float16)a3.x; a[13] = (_Float16)a3.y;
    a[14] = (_Float16)a3.z; a[15] = (_Float16)a3.w;

    // ---- B fragment (32x16 f16): lane holds N=l16; elem e -> K = k0+8*hlf+(e<8 ? e : e+8)
    const float* bp = Bm + (size_t)(k0 + hlf * 8) * N + colB;
    v16h b;
#pragma unroll
    for (int e = 0; e < 16; ++e)
      b[e] = (_Float16)bp[(size_t)(e < 8 ? e : e + 8) * N];

    acc = __builtin_amdgcn_wmma_f32_16x16x32_f16(
        /*neg_a=*/false, a, /*neg_b=*/false, b,
        /*c_mod=*/(short)0, acc, /*reuse_a=*/false, /*reuse_b=*/false);
  }

  // ---- store: VGPR r -> M = tile_m*16 + r + 8*hlf, N = colB
  float bv = 0.f;
  if (BIAS) bv = bias[colB];
  float* cp = C + (size_t)(tile_m * 16 + hlf * 8) * N + colB;
#pragma unroll
  for (int r = 0; r < 8; ++r) {
    float v = acc[r] + bv;
    if (RELU) v = fmaxf(v, 0.f);
    cp[(size_t)r * N] = v;
  }
}

// ---------------- pipeline kernels -----------------------------------------
__global__ void k_init_edges(const int* __restrict__ ei, int* src, int* dst,
                             int* mask, int E) {
  int e = blockIdx.x * blockDim.x + threadIdx.x;
  if (e < E) { src[e] = ei[e]; dst[e] = ei[E + e]; mask[e] = 1; }
}

__global__ void k_gather_embed(const int* __restrict__ x,
                               const float* __restrict__ emb,
                               float* __restrict__ h) {
  int i = blockIdx.x, t = threadIdx.x;
  h[(size_t)i * DFEAT + t] = emb[(size_t)x[i] * DFEAT + t];
}

__global__ void k_deg_init(float* deg, int N) {
  int i = blockIdx.x * blockDim.x + threadIdx.x;
  if (i < N) deg[i] = 1.0f;
}

__global__ void k_deg_acc(const int* __restrict__ dst,
                          const int* __restrict__ mask, float* deg, int E) {
  int e = blockIdx.x * blockDim.x + threadIdx.x;
  if (e < E && mask[e]) atomicAdd(&deg[dst[e]], 1.0f);
}

__global__ void k_dinv(const float* __restrict__ deg, float* dinv, int N) {
  int i = blockIdx.x * blockDim.x + threadIdx.x;
  if (i < N) dinv[i] = rsqrtf(deg[i]);
}

// out[i,f] = hw[i,f] * dinv[i]^2 + b[f]   (self loop term + bias)
__global__ void k_self_term(const float* __restrict__ hw,
                            const float* __restrict__ dinv,
                            const float* __restrict__ b,
                            float* __restrict__ out) {
  int i = blockIdx.x, t = threadIdx.x;
  float di = dinv[i];
  out[(size_t)i * DFEAT + t] = hw[(size_t)i * DFEAT + t] * di * di + b[t];
}

// out[dst,f] += dinv[src]*dinv[dst] * hw[src,f]   (one block per edge)
__global__ void k_edge_agg(const int* __restrict__ src,
                           const int* __restrict__ dst,
                           const int* __restrict__ mask,
                           const float* __restrict__ dinv,
                           const float* __restrict__ hw,
                           float* __restrict__ out) {
  int e = blockIdx.x;
  if (!mask[e]) return;
  int s = src[e], d = dst[e];
  float c = dinv[s] * dinv[d];
  int t = threadIdx.x;
  atomicAdd(&out[(size_t)d * DFEAT + t], c * hw[(size_t)s * DFEAT + t]);
}

__global__ void k_relu(float* p, int n) {
  int i = blockIdx.x * blockDim.x + threadIdx.x;
  if (i < n) { float v = p[i]; p[i] = v > 0.f ? v : 0.f; }
}

__global__ void k_wnorm(const float* __restrict__ w, float* invn) {
  __shared__ float s[DFEAT];
  int t = threadIdx.x;
  float v = w[t];
  s[t] = v * v;
  __syncthreads();
  for (int o = 64; o > 0; o >>= 1) {
    if (t < o) s[t] += s[t + o];
    __syncthreads();
  }
  if (t == 0) *invn = rsqrtf(s[0]);
}

__global__ void k_score(const float* __restrict__ h, const float* __restrict__ w,
                        const float* __restrict__ invn, float* score) {
  __shared__ float s[DFEAT];
  int i = blockIdx.x, t = threadIdx.x;
  s[t] = h[(size_t)i * DFEAT + t] * w[t];
  __syncthreads();
  for (int o = 64; o > 0; o >>= 1) {
    if (t < o) s[t] += s[t + o];
    __syncthreads();
  }
  if (t == 0) score[i] = s[0] * (*invn);
}

// Per-graph exact top-k via rank counting (stable ties, matches lax.top_k).
__global__ void k_topk(const float* __restrict__ score, int* __restrict__ newpos,
                       int* __restrict__ oldidx, float* __restrict__ tanhv,
                       int npg, int k) {
  __shared__ float s[NPG0];
  int g = blockIdx.x, t = threadIdx.x;
  float my = -__builtin_inff();
  if (t < npg) { my = score[g * npg + t]; s[t] = my; }
  __syncthreads();
  if (t < npg) {
    int rank = 0;
    for (int j = 0; j < npg; ++j) {
      float v = s[j];
      if (v > my || (v == my && j < t)) ++rank;
    }
    int old_global = g * npg + t;
    if (rank < k) {
      int nr = g * k + rank;
      newpos[old_global] = nr;
      oldidx[nr] = old_global;
      tanhv[nr]  = tanhf(my);
    } else {
      newpos[old_global] = -1;
    }
  }
}

__global__ void k_pool_gather(const float* __restrict__ h_old,
                              const int* __restrict__ oldidx,
                              const float* __restrict__ tanhv,
                              float* __restrict__ h_new) {
  int r = blockIdx.x, t = threadIdx.x;
  h_new[(size_t)r * DFEAT + t] =
      h_old[(size_t)oldidx[r] * DFEAT + t] * tanhv[r];
}

__global__ void k_edge_remap(int* src, int* dst, int* mask,
                             const int* __restrict__ newpos, int E) {
  int e = blockIdx.x * blockDim.x + threadIdx.x;
  if (e >= E) return;
  int ns = newpos[src[e]];
  int nd = newpos[dst[e]];
  int m  = mask[e] && ns >= 0 && nd >= 0;
  src[e]  = m ? ns : 0;
  dst[e]  = m ? nd : 0;
  mask[e] = m ? 1 : 0;
}

// rtot[g, 0:128]   += max over nodes,  rtot[g, 128:256] += mean over nodes
__global__ void k_readout_acc(const float* __restrict__ h, float* __restrict__ rtot,
                              int npg) {
  int g = blockIdx.x, t = threadIdx.x;
  float mx = -__builtin_inff(), sm = 0.f;
  const float* base = h + (size_t)g * npg * DFEAT + t;
  for (int r = 0; r < npg; ++r) {
    float v = base[(size_t)r * DFEAT];
    mx = fmaxf(mx, v);
    sm += v;
  }
  rtot[g * 256 + t]        += mx;
  rtot[g * 256 + 128 + t]  += sm / (float)npg;
}

__global__ void k_zero(float* p, int n) {
  int i = blockIdx.x * blockDim.x + threadIdx.x;
  if (i < n) p[i] = 0.f;
}

__global__ void k_final(const float* __restrict__ z2, const float* __restrict__ l3W,
                        const float* __restrict__ l3b, float* __restrict__ out) {
  int g = blockIdx.x * blockDim.x + threadIdx.x;
  if (g >= BGR) return;
  float acc = l3b[0];
  for (int j = 0; j < 64; ++j) acc += z2[g * 64 + j] * l3W[j];
  out[g] = 1.f / (1.f + expf(-acc));
}

// ---------------- host orchestration ----------------------------------------
extern "C" void kernel_launch(void* const* d_in, const int* in_sizes, int n_in,
                              void* d_out, int out_size, void* d_ws, size_t ws_size,
                              hipStream_t stream) {
  const int*   x    = (const int*)d_in[0];
  const int*   ei   = (const int*)d_in[1];
  const float* emb  = (const float*)d_in[3];
  const float* Wl[3]  = {(const float*)d_in[4], (const float*)d_in[7],  (const float*)d_in[10]};
  const float* bl[3]  = {(const float*)d_in[5], (const float*)d_in[8],  (const float*)d_in[11]};
  const float* pwl[3] = {(const float*)d_in[6], (const float*)d_in[9],  (const float*)d_in[12]};
  const float* l1W = (const float*)d_in[13];
  const float* l1b = (const float*)d_in[14];
  const float* l2W = (const float*)d_in[15];
  const float* l2b = (const float*)d_in[16];
  const float* l3W = (const float*)d_in[17];
  const float* l3b = (const float*)d_in[18];
  float* out = (float*)d_out;

  // workspace carve-up (deterministic)
  char* wp = (char*)d_ws;
  auto carve = [&](size_t bytes) -> void* {
    void* p = (void*)wp;
    wp += (bytes + 255) & ~(size_t)255;
    return p;
  };
  float* hA    = (float*)carve((size_t)N0 * DFEAT * 4);   // ping
  float* hB    = (float*)carve((size_t)N0 * DFEAT * 4);   // pong
  float* hw    = (float*)carve((size_t)N0 * DFEAT * 4);   // GEMM output
  float* deg   = (float*)carve((size_t)N0 * 4);
  float* dinv  = (float*)carve((size_t)N0 * 4);
  float* score = (float*)carve((size_t)N0 * 4);
  int*   newpos= (int*)  carve((size_t)N0 * 4);
  int*   oldidx= (int*)  carve((size_t)N0 * 4);
  float* tanhv = (float*)carve((size_t)N0 * 4);
  int*   esrc  = (int*)  carve((size_t)NEDGE * 4);
  int*   edst  = (int*)  carve((size_t)NEDGE * 4);
  int*   emask = (int*)  carve((size_t)NEDGE * 4);
  float* rtot  = (float*)carve((size_t)BGR * 256 * 4);
  float* z1    = (float*)carve((size_t)BGR * 128 * 4);
  float* z2    = (float*)carve((size_t)BGR * 64 * 4);
  float* invn  = (float*)carve(256);

  const int npgs[4] = {128, 103, 83, 67};  // npg in / pooled sizes per layer

  // init
  k_init_edges<<<(NEDGE + 255) / 256, 256, 0, stream>>>(ei, esrc, edst, emask, NEDGE);
  k_gather_embed<<<N0, DFEAT, 0, stream>>>(x, emb, hA);
  k_zero<<<(BGR * 256 + 255) / 256, 256, 0, stream>>>(rtot, BGR * 256);

  for (int L = 0; L < 3; ++L) {
    const int npg  = npgs[L];
    const int kk   = npgs[L + 1];
    const int Nin  = BGR * npg;   // multiple of 16 (32768 / 26368 / 21248)
    const int Nnew = BGR * kk;

    // hw = h @ W_L   (WMMA, K=128, no bias/relu)
    k_wmma_gemm<128, false, false><<<dim3(Nin / 16, DFEAT / 16), 32, 0, stream>>>(
        hA, Wl[L], hw, DFEAT, nullptr);

    // GCN normalization + aggregation into hB
    k_deg_init<<<(Nin + 255) / 256, 256, 0, stream>>>(deg, Nin);
    k_deg_acc<<<(NEDGE + 255) / 256, 256, 0, stream>>>(edst, emask, deg, NEDGE);
    k_dinv<<<(Nin + 255) / 256, 256, 0, stream>>>(deg, dinv, Nin);
    k_self_term<<<Nin, DFEAT, 0, stream>>>(hw, dinv, bl[L], hB);
    k_edge_agg<<<NEDGE, DFEAT, 0, stream>>>(esrc, edst, emask, dinv, hw, hB);
    k_relu<<<((size_t)Nin * DFEAT + 255) / 256, 256, 0, stream>>>(hB, Nin * DFEAT);

    // top-k pool: hB -> hA (Nnew rows)
    k_wnorm<<<1, DFEAT, 0, stream>>>(pwl[L], invn);
    k_score<<<Nin, DFEAT, 0, stream>>>(hB, pwl[L], invn, score);
    k_topk<<<BGR, DFEAT, 0, stream>>>(score, newpos, oldidx, tanhv, npg, kk);
    k_pool_gather<<<Nnew, DFEAT, 0, stream>>>(hB, oldidx, tanhv, hA);
    k_edge_remap<<<(NEDGE + 255) / 256, 256, 0, stream>>>(esrc, edst, emask, newpos, NEDGE);

    // readout on pooled features
    k_readout_acc<<<BGR, DFEAT, 0, stream>>>(hA, rtot, kk);
  }

  // MLP head: z1 = relu(rtot @ l1W + l1b), z2 = relu(z1 @ l2W + l2b)
  k_wmma_gemm<256, true, true><<<dim3(BGR / 16, 128 / 16), 32, 0, stream>>>(
      rtot, l1W, z1, 128, l1b);
  k_wmma_gemm<128, true, true><<<dim3(BGR / 16, 64 / 16), 32, 0, stream>>>(
      z1, l2W, z2, 64, l2b);
  k_final<<<1, BGR, 0, stream>>>(z2, l3W, l3b, out);
}
